// MultiHeadAttention_73504070303932
// MI455X (gfx1250) — matrix operand order repaired and checked by reading, not compiled, not run on hardware
//
#include <hip/hip_runtime.h>

// ---------------------------------------------------------------------------
// MI455X (gfx1250, wave32) multi-head attention forward.
// Matmuls via v_wmma_f32_16x16x32_bf16; contiguous LDS staging via the CDNA5
// async direct-to-LDS path (global_load_async_to_lds_b128 + ASYNCcnt).
// ---------------------------------------------------------------------------

typedef __bf16 bf16_t;
typedef __attribute__((ext_vector_type(4)))  __bf16 bf16x4;
typedef __attribute__((ext_vector_type(8)))  __bf16 bf16x8;
typedef __attribute__((ext_vector_type(16))) __bf16 bf16x16;
typedef __attribute__((ext_vector_type(8)))  float  f32x8;

#define SEQ    1024
#define MODEL  1024
#define HEADS  16
#define VDIM   64
#define BATCH  4
#define ROWS   (BATCH * SEQ)   // 4096 tokens total

__device__ __forceinline__ f32x8 wmma_bf16(bf16x16 a, bf16x16 b, f32x8 c) {
  // (neg_a, A, neg_b, B, c_mod, C, reuse_a, reuse_b)
  return __builtin_amdgcn_wmma_f32_16x16x32_bf16(false, a, false, b, (short)0, c,
                                                 false, false);
}

// Async memory->LDS copy, 16B per lane. VDST = 32-bit LDS byte address,
// VADDR = 64-bit global address (GV mode, SADDR=off). Tracked by ASYNCcnt.
__device__ __forceinline__ void async_copy_b128(void* lds, const void* gptr) {
  unsigned lds_off = (unsigned)(size_t)lds;  // generic addr low 32b == LDS offset
  asm volatile("global_load_async_to_lds_b128 %0, %1, off"
               :: "v"(lds_off), "v"(gptr) : "memory");
}
__device__ __forceinline__ void wait_async0() {
#if __has_builtin(__builtin_amdgcn_s_wait_asynccnt)
  __builtin_amdgcn_s_wait_asynccnt(0);
#else
  asm volatile("s_wait_asynccnt 0x0" ::: "memory");
#endif
}

__device__ __forceinline__ bf16x16 cat8(bf16x8 lo, bf16x8 hi) {
  bf16x16 r;
#pragma unroll
  for (int i = 0; i < 8; ++i) { r[i] = lo[i]; r[i + 8] = hi[i]; }
  return r;
}

// A-operand fragment (16x32 bf16) from a row-major [m][k] LDS tile.
// Lane (m = lane&15): elems 0..7 -> k = kb..kb+7, elems 8..15 -> k = kb+16..kb+23
// where kb = (lane&16) ? 8 : 0 (+ caller k-step offset). ISA 7.12.2 layout.
__device__ __forceinline__ bf16x16 frag_a(const bf16_t* base, int ld, int m, int kb) {
  const bf16_t* p = base + m * ld + kb;
  return cat8(*(const bf16x8*)p, *(const bf16x8*)(p + 16));
}

// B-operand fragment (32x16 bf16) from a [n][k] (k-contiguous) LDS tile.
// Lane (n = lane&15): elems 0..7 -> k = hb..hb+7, elems 8..15 -> k = hb+16..hb+23,
// hb = (lane&16) ? 8 : 0 (+ k-step); caller passes kb = (lane&16) ? 16 : 0 (+ k-step).
__device__ __forceinline__ bf16x16 frag_b(const bf16_t* base, int ld, int n, int kb) {
  const bf16_t* p = base + n * ld + (kb - (kb & 16) + ((kb & 16) >> 1));
  return cat8(*(const bf16x8*)p, *(const bf16x8*)(p + 16));
}

// Reductions across the 16 lanes of a half-wave (rows of a WMMA C tile live on
// lanes 0-15 (m=r) and 16-31 (m=r+8) independently).
__device__ __forceinline__ float half16_max(float v) {
#pragma unroll
  for (int m = 1; m < 16; m <<= 1) v = fmaxf(v, __shfl_xor(v, m, 32));
  return v;
}
__device__ __forceinline__ float half16_sum(float v) {
#pragma unroll
  for (int m = 1; m < 16; m <<= 1) v += __shfl_xor(v, m, 32);
  return v;
}

// ---------------------------------------------------------------------------
// f32 -> bf16 conversion (4 elems / thread)
// ---------------------------------------------------------------------------
__global__ void cvt_f32_bf16(const float* __restrict__ in,
                             bf16_t* __restrict__ out, int n) {
  int i = (blockIdx.x * blockDim.x + threadIdx.x) * 4;
  if (i < n) {
    float4 v = *(const float4*)(in + i);
    bf16x4 o = {(bf16_t)v.x, (bf16_t)v.y, (bf16_t)v.z, (bf16_t)v.w};
    *(bf16x4*)(out + i) = o;
  }
}

// ---------------------------------------------------------------------------
// Shared GEMM core: C(128x64 f32) = A(row-major, ld=MODEL) * B(row-major, ld=MODEL)
// 256 threads = 8 waves in a 4x2 grid; each wave owns a 32x32 patch (2x2 WMMAs).
// A staged row-major [128][32] via async-to-LDS; B staged transposed [64][32]
// through VGPRs so both fragment loads are contiguous, aligned ds_load_b128s.
// ---------------------------------------------------------------------------
__device__ __forceinline__ void gemm_tile_128x64(
    const bf16_t* __restrict__ A, const bf16_t* __restrict__ Bm,
    int row0, int col0, bf16_t (*As)[40], bf16_t (*Bs)[40], f32x8 acc[2][2]) {
  const int tid  = threadIdx.x;
  const int lane = tid & 31;
  const int wv   = tid >> 5;
  const int wm   = wv >> 1, wn = wv & 1;
  const int n15  = lane & 15;
  const int akb  = (lane & 16) ? 8 : 0;
  const int bkb  = (lane & 16) ? 16 : 0;
  const int ar = tid >> 1, ac = (tid & 1) * 16;  // A stage: 16 halfs / thread
  const int br = tid >> 3, bc = (tid & 7) * 8;   // B stage: 8 halfs / thread

  const f32x8 z = {0.f, 0.f, 0.f, 0.f, 0.f, 0.f, 0.f, 0.f};
  acc[0][0] = z; acc[0][1] = z; acc[1][0] = z; acc[1][1] = z;

  for (int k0 = 0; k0 < MODEL; k0 += 32) {
    __syncthreads();  // previous tile fully consumed
    {
      const bf16_t* ga = A + (size_t)(row0 + ar) * MODEL + k0 + ac;
      async_copy_b128(&As[ar][ac], ga);          // memory -> LDS, no VGPRs
      async_copy_b128(&As[ar][ac + 8], ga + 8);
      bf16x8 wrow = *(const bf16x8*)(Bm + (size_t)(k0 + br) * MODEL + col0 + bc);
#pragma unroll
      for (int i = 0; i < 8; ++i) Bs[bc + i][br] = wrow[i];  // transpose into LDS
      if (k0 + 32 < MODEL)  // prefetch next A tile into near caches (WGP scope)
        __builtin_prefetch(ga + 32, 0, 3);
    }
    wait_async0();
    __syncthreads();

    bf16x16 a0 = frag_a(&As[0][0], 40, wm * 32 + n15, akb);
    bf16x16 a1 = frag_a(&As[0][0], 40, wm * 32 + 16 + n15, akb);
    bf16x16 b0 = frag_b(&Bs[0][0], 40, wn * 32 + n15, bkb);
    bf16x16 b1 = frag_b(&Bs[0][0], 40, wn * 32 + 16 + n15, bkb);
    acc[0][0] = wmma_bf16(a0, b0, acc[0][0]);
    acc[0][1] = wmma_bf16(a0, b1, acc[0][1]);
    acc[1][0] = wmma_bf16(a1, b0, acc[1][0]);
    acc[1][1] = wmma_bf16(a1, b1, acc[1][1]);
  }
}

// ---------------------------------------------------------------------------
// Projections: z==0 -> Q = (x@Wq)/sqrt(d)   (bf16, [b,h,l,v])
//              z==1 -> KV = x@Wkv, K = KV*(1+k_weights[h,v])  (bf16, [b,h,l,v])
// ---------------------------------------------------------------------------
__global__ void __launch_bounds__(256)
proj_kernel(const bf16_t* __restrict__ Xb, const bf16_t* __restrict__ Wqb,
            const bf16_t* __restrict__ Wkvb, const float* __restrict__ kw,
            bf16_t* __restrict__ Qb, bf16_t* __restrict__ Kb,
            bf16_t* __restrict__ KVb) {
  __shared__ bf16_t As[128][40];
  __shared__ bf16_t Bs[64][40];
  f32x8 acc[2][2];
  const int row0 = blockIdx.y * 128, col0 = blockIdx.x * 64;
  const bf16_t* W = blockIdx.z ? Wkvb : Wqb;
  gemm_tile_128x64(Xb, W, row0, col0, As, Bs, acc);

  const int lane = threadIdx.x & 31;
  const int wv = threadIdx.x >> 5;
  const int wm = wv >> 1, wn = wv & 1;
  const int n15 = lane & 15, mofs = (lane & 16) ? 8 : 0;
#pragma unroll
  for (int mt = 0; mt < 2; ++mt)
#pragma unroll
    for (int nt = 0; nt < 2; ++nt)
#pragma unroll
      for (int r = 0; r < 8; ++r) {
        int gr = row0 + wm * 32 + mt * 16 + mofs + r;   // token index
        int gc = col0 + wn * 32 + nt * 16 + n15;        // h*64 + v
        int b = gr >> 10, l = gr & 1023, h = gc >> 6, vd = gc & 63;
        size_t oi = ((size_t)(b * HEADS + h) * SEQ + l) * VDIM + vd;
        float v = acc[mt][nt][r];
        if (blockIdx.z == 0) {
          Qb[oi] = (bf16_t)(v * 0.03125f);  // 1/sqrt(1024)
        } else {
          KVb[oi] = (bf16_t)v;
          Kb[oi]  = (bf16_t)(v * (1.0f + kw[h * VDIM + vd]));
        }
      }
}

// ---------------------------------------------------------------------------
// Flash-style causal attention per (b,h). Block = 4 waves x 16 query rows.
// Iterates K/V tiles of 32 keys; online softmax; both matmuls via WMMA.
// ---------------------------------------------------------------------------
__global__ void __launch_bounds__(128)
attn_kernel(const bf16_t* __restrict__ Qb, const bf16_t* __restrict__ Kb,
            const bf16_t* __restrict__ Vb, bf16_t* __restrict__ Yb) {
  __shared__ bf16_t Qs[64][72];      // Q tile, row-major [m][k]
  __shared__ bf16_t Ks[32][72];      // K tile, row-major [n][k] (B operand of QK^T)
  __shared__ bf16_t Vs[64][40];      // V^T tile, [n][k] (B operand of P@V)
  __shared__ bf16_t Ps[4][16][40];   // per-wave P strip, [m][k] (A operand)

  const int tid = threadIdx.x;
  const int lane = tid & 31;
  const int wv = tid >> 5;
  const int n15 = lane & 15;
  const int mofs = (lane & 16) ? 8 : 0;
  const int akb = (lane & 16) ? 8 : 0;
  const int bkb = (lane & 16) ? 16 : 0;
  const int bh = blockIdx.y;
  const int h = bh & (HEADS - 1);
  const int bz = bh >> 4;
  const int q0 = blockIdx.x * 64;
  const int srow = q0 + wv * 16;     // this wave's first query row (global)

  {  // stage Q tile (64x64) via async-to-LDS
    int row = tid >> 1, c0 = (tid & 1) * 32;
    const bf16_t* src = Qb + ((size_t)bh * SEQ + q0 + row) * VDIM + c0;
#pragma unroll
    for (int j = 0; j < 4; ++j)
      async_copy_b128(&Qs[row][c0 + j * 8], src + j * 8);
  }
  wait_async0();
  __syncthreads();

  // Q operand fragments (k = 0..31 and 32..63), resident for the whole block
  bf16x16 aq0 = frag_a(&Qs[wv * 16][0], 72, n15, akb);
  bf16x16 aq1 = frag_a(&Qs[wv * 16][0], 72, n15, 32 + akb);

  const f32x8 z = {0.f, 0.f, 0.f, 0.f, 0.f, 0.f, 0.f, 0.f};
  f32x8 o[4] = {z, z, z, z};         // O accumulator, 16x64
  float mrow[8], lrow[8];
#pragma unroll
  for (int r = 0; r < 8; ++r) { mrow[r] = -3.0e38f; lrow[r] = 0.f; }

  const int jend = q0 + 64;          // causal: keys < end of this query block
  for (int jk = 0; jk < jend; jk += 32) {
    __syncthreads();                 // previous K/V tile fully consumed
    {  // stage K tile row-major (async) and V tile transposed (VGPR path)
      int row = tid >> 2, c0 = (tid & 3) * 16;
      const bf16_t* ks = Kb + ((size_t)bh * SEQ + jk + row) * VDIM + c0;
      async_copy_b128(&Ks[row][c0], ks);
      async_copy_b128(&Ks[row][c0 + 8], ks + 8);
      const bf16_t* vsrc = Vb + ((size_t)bh * SEQ + jk + row) * VDIM + c0;
      bf16x16 vv = cat8(*(const bf16x8*)vsrc, *(const bf16x8*)(vsrc + 8));
#pragma unroll
      for (int i = 0; i < 16; ++i) Vs[c0 + i][row] = vv[i];
    }
    wait_async0();
    __syncthreads();

    // S(16x32) = Qstrip(16x64) @ K^T : two k-steps, two n-tiles
    f32x8 s0 = z, s1 = z;
    {
      bf16x16 bk00 = frag_b(&Ks[0][0], 72, n15,      bkb);
      bf16x16 bk01 = frag_b(&Ks[0][0], 72, n15,      32 + bkb);
      bf16x16 bk10 = frag_b(&Ks[0][0], 72, 16 + n15, bkb);
      bf16x16 bk11 = frag_b(&Ks[0][0], 72, 16 + n15, 32 + bkb);
      s0 = wmma_bf16(aq0, bk00, s0);
      s0 = wmma_bf16(aq1, bk01, s0);
      s1 = wmma_bf16(aq0, bk10, s1);
      s1 = wmma_bf16(aq1, bk11, s1);
    }

    // causal mask + online softmax (row m lives on one 16-lane half-wave)
#pragma unroll
    for (int r = 0; r < 8; ++r) {
      int grow = srow + mofs + r;
      float v0 = s0[r], v1 = s1[r];
      if (jk + n15 > grow)      v0 = -3.0e38f;
      if (jk + 16 + n15 > grow) v1 = -3.0e38f;
      float tm   = half16_max(fmaxf(v0, v1));
      float newm = fmaxf(mrow[r], tm);
      float sc   = __expf(mrow[r] - newm);
      float p0   = __expf(v0 - newm);
      float p1   = __expf(v1 - newm);
      float ps   = half16_sum(p0 + p1);
      lrow[r] = lrow[r] * sc + ps;
      mrow[r] = newm;
      o[0][r] *= sc; o[1][r] *= sc; o[2][r] *= sc; o[3][r] *= sc;
      // re-layout P as an A operand via per-wave-private LDS
      Ps[wv][mofs + r][n15]      = (bf16_t)p0;
      Ps[wv][mofs + r][16 + n15] = (bf16_t)p1;
    }

    // O += P(16x32) @ V(32x64). Intra-wave DS ops are in-order (DScnt rules),
    // so the wave's own P stores are visible to its fragment loads.
    bf16x16 ap = frag_a(&Ps[wv][0][0], 40, n15, akb);
#pragma unroll
    for (int nt = 0; nt < 4; ++nt) {
      bf16x16 bv = frag_b(&Vs[0][0], 40, nt * 16 + n15, bkb);
      o[nt] = wmma_bf16(ap, bv, o[nt]);
    }
  }

  // normalize and emit Y as [b, l, h*64+v] bf16 for the output GEMM
#pragma unroll
  for (int nt = 0; nt < 4; ++nt)
#pragma unroll
    for (int r = 0; r < 8; ++r) {
      int grow = srow + mofs + r;
      float val = o[nt][r] / lrow[r];
      Yb[((size_t)bz * SEQ + grow) * (HEADS * VDIM) + h * VDIM + nt * 16 + n15] =
          (bf16_t)val;
    }
}

// ---------------------------------------------------------------------------
// out = Y @ Wo  (f32 result)
// ---------------------------------------------------------------------------
__global__ void __launch_bounds__(256)
out_gemm_kernel(const bf16_t* __restrict__ Yb, const bf16_t* __restrict__ Wob,
                float* __restrict__ Out) {
  __shared__ bf16_t As[128][40];
  __shared__ bf16_t Bs[64][40];
  f32x8 acc[2][2];
  const int row0 = blockIdx.y * 128, col0 = blockIdx.x * 64;
  gemm_tile_128x64(Yb, Wob, row0, col0, As, Bs, acc);

  const int lane = threadIdx.x & 31;
  const int wv = threadIdx.x >> 5;
  const int wm = wv >> 1, wn = wv & 1;
  const int n15 = lane & 15, mofs = (lane & 16) ? 8 : 0;
#pragma unroll
  for (int mt = 0; mt < 2; ++mt)
#pragma unroll
    for (int nt = 0; nt < 2; ++nt)
#pragma unroll
      for (int r = 0; r < 8; ++r) {
        int gr = row0 + wm * 32 + mt * 16 + mofs + r;
        int gc = col0 + wn * 32 + nt * 16 + n15;
        Out[(size_t)gr * MODEL + gc] = acc[mt][nt][r];
      }
}

// ---------------------------------------------------------------------------
extern "C" void kernel_launch(void* const* d_in, const int* in_sizes, int n_in,
                              void* d_out, int out_size, void* d_ws, size_t ws_size,
                              hipStream_t stream) {
  const float* x    = (const float*)d_in[0];   // (4,1024,1024)
  const float* w_q  = (const float*)d_in[1];   // (1024,1024)
  const float* w_kv = (const float*)d_in[2];   // (1024,1024)
  const float* w_o  = (const float*)d_in[3];   // (1024,1024)
  const float* kw   = (const float*)d_in[4];   // (1,16,1,64)
  float* out = (float*)d_out;

  char* ws = (char*)d_ws;                      // 46 MiB used
  bf16_t* Xb   = (bf16_t*)(ws);                // 8 MiB  [4096][1024]
  bf16_t* Wqb  = (bf16_t*)(ws + (8ull  << 20));
  bf16_t* Wkvb = (bf16_t*)(ws + (10ull << 20));
  bf16_t* Wob  = (bf16_t*)(ws + (12ull << 20));
  bf16_t* Qb   = (bf16_t*)(ws + (14ull << 20)); // [64][1024][64]
  bf16_t* Kb   = (bf16_t*)(ws + (22ull << 20));
  bf16_t* KVb  = (bf16_t*)(ws + (30ull << 20));
  bf16_t* Yb   = (bf16_t*)(ws + (38ull << 20)); // [4096][1024]

  cvt_f32_bf16<<<ROWS * MODEL / 1024, 256, 0, stream>>>(x, Xb, ROWS * MODEL);
  cvt_f32_bf16<<<MODEL * MODEL / 1024, 256, 0, stream>>>(w_q, Wqb, MODEL * MODEL);
  cvt_f32_bf16<<<MODEL * MODEL / 1024, 256, 0, stream>>>(w_kv, Wkvb, MODEL * MODEL);
  cvt_f32_bf16<<<MODEL * MODEL / 1024, 256, 0, stream>>>(w_o, Wob, MODEL * MODEL);

  proj_kernel<<<dim3(MODEL / 64, ROWS / 128, 2), 256, 0, stream>>>(
      Xb, Wqb, Wkvb, kw, Qb, Kb, KVb);

  attn_kernel<<<dim3(SEQ / 64, BATCH * HEADS), 128, 0, stream>>>(Qb, Kb, KVb, Yb);

  out_gemm_kernel<<<dim3(MODEL / 64, ROWS / 128), 256, 0, stream>>>(Yb, Wob, out);
}
